// WorldModel_5111011082546
// MI455X (gfx1250) — matrix operand order repaired and checked by reading, not compiled, not run on hardware
//
#include <hip/hip_runtime.h>
#include <hip/hip_bf16.h>
#include <math.h>

typedef __attribute__((ext_vector_type(16))) _Float16 v16h;
typedef __attribute__((ext_vector_type(8)))  _Float16 h8;
typedef __attribute__((ext_vector_type(8)))  float    v8f;

#define B_    2
#define S_    2048
#define D_    512
#define L_    4
#define H_    8
#define DH_   64
#define DI_   512      // H*DH
#define NQKV_ 1536
#define DHID_ 1365
#define NFF_  2730
#define NFFP_ 2752     // NFF padded to 43*64 (zero pad)
#define KFFP_ 1376     // DHID padded to mult of 32 (zero pad)
#define M_    4096     // B*S

union V16 { v16h v; h8 h[2]; };

// -------- CDNA5 async global->LDS copy (ASYNCcnt-tracked, bypasses VGPRs) ---
__device__ __forceinline__ void async_copy16(unsigned lds_addr, const void* gptr) {
  asm volatile("global_load_async_to_lds_b128 %0, %1, off"
               :: "v"(lds_addr), "v"((unsigned long long)(size_t)gptr)
               : "memory");
}
__device__ __forceinline__ void async_wait0() {
  asm volatile("s_wait_asynccnt 0" ::: "memory");
}
__device__ __forceinline__ unsigned lds_off(const void* p) {
  return (unsigned)(size_t)p;   // low 32 bits of generic addr = LDS byte address
}

// ---------------------------------------------------------------- rmsnorm ----
__global__ __launch_bounds__(256)
void rmsnorm_kernel(const float* __restrict__ src, const float* __restrict__ w,
                    float* __restrict__ dstF, _Float16* __restrict__ dstH) {
  int row = blockIdx.x;
  const float* r = src + (size_t)row * D_;
  __shared__ float red[8];
  float a0 = r[threadIdx.x];
  float a1 = r[threadIdx.x + 256];
  float ss = a0 * a0 + a1 * a1;
  #pragma unroll
  for (int off = 1; off < 32; off <<= 1) ss += __shfl_xor(ss, off, 32);
  if ((threadIdx.x & 31) == 0) red[threadIdx.x >> 5] = ss;
  __syncthreads();
  float tot = 0.f;
  #pragma unroll
  for (int i = 0; i < 8; ++i) tot += red[i];
  float inv = rsqrtf(tot * (1.0f / (float)D_) + 1.1920929e-07f);
  float o0 = a0 * inv * w[threadIdx.x];
  float o1 = a1 * inv * w[threadIdx.x + 256];
  if (dstH) {
    dstH[(size_t)row * D_ + threadIdx.x]       = (_Float16)o0;
    dstH[(size_t)row * D_ + threadIdx.x + 256] = (_Float16)o1;
  } else {
    dstF[(size_t)row * D_ + threadIdx.x]       = o0;
    dstF[(size_t)row * D_ + threadIdx.x + 256] = o1;
  }
}

// -------------------------------------------- weight convert + transpose ----
// Wt[n*Kpad + k] = (k<K && n<N) ? (f16)W[k*N + n] : 0   (one-shot per launch)
__global__ __launch_bounds__(256)
void wcvt_kernel(const float* __restrict__ W, _Float16* __restrict__ Wt,
                 int K, int N, int Kpad, int Npad) {
  long long idx = (long long)blockIdx.x * 256 + threadIdx.x;
  if (idx >= (long long)Npad * Kpad) return;
  int n = (int)(idx / Kpad);
  int k = (int)(idx % Kpad);
  float v = (k < K && n < N) ? W[(size_t)k * N + n] : 0.f;
  Wt[(size_t)n * Kpad + k] = (_Float16)v;
}

__global__ __launch_bounds__(256)
void padb_kernel(const float* __restrict__ b, float* __restrict__ bp,
                 int n, int npad) {
  int i = blockIdx.x * 256 + threadIdx.x;
  if (i < npad) bp[i] = (i < n) ? b[i] : 0.f;
}

// ------------------------------------------------------------- WMMA GEMM ----
// C[M,N](f32) = A[M,K](f16 row-major) @ Wt[N,K](f16, pre-transposed) (+bias)
// Requires K%32==0, N%64==0 (callers pad with zeros). BM=128 BN=64 BK=32,
// 256 threads = 8 waves, 32x32 per wave. Double-buffered async b128 staging:
// tile ks+1 copies overlap with tile ks WMMA work.
__global__ __launch_bounds__(256)
void gemm_wmma(const _Float16* __restrict__ A, const _Float16* __restrict__ Wt,
               const float* __restrict__ bias, float* __restrict__ C,
               int K, int N, int accumulate) {
  __shared__ __attribute__((aligned(16))) _Float16 As[2][128][40];
  __shared__ __attribute__((aligned(16))) _Float16 Bs[2][64][40];   // [n][k]
  const int tid  = threadIdx.x;
  const int lane = tid & 31;
  const int wave = tid >> 5;
  const int waveRow = wave >> 1;   // 0..3
  const int waveCol = wave & 1;    // 0..1
  const int m0 = blockIdx.y * 128;
  const int n0 = blockIdx.x * 64;
  const int lm = lane & 15;
  const int lh = lane >> 4;

  v8f c[2][2] = {};

  // copy mapping: 16B per async op; A: 128 rows x 4 chunks (2 units/thread),
  // B: 64 rows x 4 chunks (1 unit/thread)
  const int ar = tid >> 2;          // 0..63
  const int ac = (tid & 3) * 8;     // halves
  unsigned ldsA0[2], ldsA1[2], ldsB[2];
  #pragma unroll
  for (int bf = 0; bf < 2; ++bf) {
    ldsA0[bf] = lds_off(&As[bf][ar][ac]);
    ldsA1[bf] = lds_off(&As[bf][ar + 64][ac]);
    ldsB[bf]  = lds_off(&Bs[bf][ar][ac]);
  }
  const _Float16* ga0 = A  + (size_t)(m0 + ar) * K + ac;
  const _Float16* ga1 = A  + (size_t)(m0 + ar + 64) * K + ac;
  const _Float16* gb  = Wt + (size_t)(n0 + ar) * K + ac;

  const int nk = K >> 5;
  // prologue: issue tile 0 into buffer 0
  async_copy16(ldsA0[0], ga0);
  async_copy16(ldsA1[0], ga1);
  async_copy16(ldsB[0],  gb);

  for (int ks = 0; ks < nk; ++ks) {
    const int buf = ks & 1;
    async_wait0();       // this wave's copies for tile ks have landed
    __syncthreads();     // all waves landed; buf^1 readers finished last iter
    if (ks + 1 < nk) {   // overlap tile ks+1 copies with tile ks compute
      const int k1 = (ks + 1) << 5;
      async_copy16(ldsA0[buf ^ 1], ga0 + k1);
      async_copy16(ldsA1[buf ^ 1], ga1 + k1);
      async_copy16(ldsB[buf ^ 1],  gb  + k1);
    }

    v16h afrag[2], bfrag[2];
    #pragma unroll
    for (int tm = 0; tm < 2; ++tm) {
      int row = waveRow * 32 + tm * 16 + lm;
      V16 u;
      u.h[0] = *(const h8*)&As[buf][row][8 * lh];
      u.h[1] = *(const h8*)&As[buf][row][16 + 8 * lh];
      afrag[tm] = u.v;
    }
    #pragma unroll
    for (int tn = 0; tn < 2; ++tn) {
      int col = waveCol * 32 + tn * 16 + lm;
      V16 u;
      u.h[0] = *(const h8*)&Bs[buf][col][8 * lh];
      u.h[1] = *(const h8*)&Bs[buf][col][16 + 8 * lh];
      bfrag[tn] = u.v;
    }
    #pragma unroll
    for (int tm = 0; tm < 2; ++tm)
      #pragma unroll
      for (int tn = 0; tn < 2; ++tn)
        c[tm][tn] = __builtin_amdgcn_wmma_f32_16x16x32_f16(
            false, afrag[tm], false, bfrag[tn], (short)0, c[tm][tn], false, false);
    __syncthreads();     // everyone done reading buf before it is refilled
  }

  #pragma unroll
  for (int tm = 0; tm < 2; ++tm) {
    #pragma unroll
    for (int tn = 0; tn < 2; ++tn) {
      const int col = n0 + waveCol * 32 + tn * 16 + lm;
      const float bv = bias ? bias[col] : 0.f;
      #pragma unroll
      for (int r = 0; r < 8; ++r) {
        const int row = m0 + waveRow * 32 + tm * 16 + lh * 8 + r;
        const size_t idx = (size_t)row * N + col;
        float v = c[tm][tn][r] + bv;
        if (accumulate) v += C[idx];
        C[idx] = v;
      }
    }
  }
}

// ---------------------------------------------------------------- mix gate --
__global__ __launch_bounds__(256)
void mix_kernel(const _Float16* __restrict__ xh, const float* __restrict__ mw,
                const float* __restrict__ mb, float* __restrict__ mix) {
  int idx = blockIdx.x * 256 + threadIdx.x;       // row*H + h
  if (idx >= M_ * H_) return;
  int row = idx >> 3;
  int h   = idx & 7;
  const _Float16* xr = xh + (size_t)row * D_;
  float acc = 0.f;
  for (int k = 0; k < D_; ++k) acc += (float)xr[k] * mw[k * H_ + h];
  acc += mb[h];
  mix[idx] = 1.f / (1.f + __expf(-acc));
}

// -------------------------------------------------------------- GEGLU ------
// g_h[row][j] (stride KFFP_, zero-padded) = a * gelu_exact(g)
__global__ __launch_bounds__(256)
void glu_kernel(const float* __restrict__ hbuf, _Float16* __restrict__ g) {
  long long idx = (long long)blockIdx.x * 256 + threadIdx.x;
  if (idx >= (long long)M_ * KFFP_) return;
  int row = (int)(idx / KFFP_);
  int j   = (int)(idx % KFFP_);
  float val = 0.f;
  if (j < DHID_) {
    float a  = hbuf[(size_t)row * NFFP_ + j];
    float gg = hbuf[(size_t)row * NFFP_ + DHID_ + j];
    val = a * (0.5f * gg * (1.f + erff(gg * 0.70710678118654752f)));
  }
  g[(size_t)row * KFFP_ + j] = (_Float16)val;
}

// -------------------------------------------------- flash block-causal attn -
// grid (S/64, H, B), 128 threads = 4 waves, 16 queries per wave, Dh=64.
#define LN10000_OVER_32 0.28782313662425572f
__global__ __launch_bounds__(128)
void attn_kernel(const float* __restrict__ qkv, const float* __restrict__ mix,
                 float* __restrict__ firstv, int storeFirst, int useMix,
                 _Float16* __restrict__ o) {
  __shared__ __attribute__((aligned(16))) _Float16 Qs[64][72];
  __shared__ __attribute__((aligned(16))) _Float16 Ks[64][72];
  __shared__ __attribute__((aligned(16))) _Float16 Vt[64][72];       // [dh][key]
  __shared__ __attribute__((aligned(16))) _Float16 Ps[4][16][72];    // per-wave P

  const int qt   = blockIdx.x;
  const int h    = blockIdx.y;
  const int b    = blockIdx.z;
  const int tid  = threadIdx.x;
  const int lane = tid & 31;
  const int wave = tid >> 5;
  const int q0   = qt * 64;
  const size_t rowBase = (size_t)b * S_;

  // stage Q tile with RoPE + scale (0.125 = Dh^-0.5)
  {
    const int r  = tid >> 1;
    const int i0 = (tid & 1) * 16;
    const int pos = q0 + r;
    const float* qp = qkv + (rowBase + pos) * NQKV_ + h * DH_;
    #pragma unroll
    for (int ii = 0; ii < 16; ++ii) {
      int i = i0 + ii;
      float inv = __expf(-LN10000_OVER_32 * (float)i);
      float ang = (float)pos * inv;
      float cs = __cosf(ang), sn = __sinf(ang);
      float t1 = qp[2 * i], t2 = qp[2 * i + 1];
      Qs[r][2 * i]     = (_Float16)((t1 * cs - t2 * sn) * 0.125f);
      Qs[r][2 * i + 1] = (_Float16)((t1 * sn + t2 * cs) * 0.125f);
    }
  }
  __syncthreads();

  const int lm = lane & 15;
  const int lh = lane >> 4;

  v16h aQ[2];
  #pragma unroll
  for (int kc = 0; kc < 2; ++kc) {
    V16 u;
    u.h[0] = *(const h8*)&Qs[wave * 16 + lm][kc * 32 + 8 * lh];
    u.h[1] = *(const h8*)&Qs[wave * 16 + lm][kc * 32 + 16 + 8 * lh];
    aQ[kc] = u.v;
  }

  float mrow[8], lrow[8];
  v8f acc[4] = {};
  #pragma unroll
  for (int r = 0; r < 8; ++r) { mrow[r] = -3.0e38f; lrow[r] = 0.f; }

  for (int kt = 0; kt <= qt; ++kt) {
    __syncthreads();
    // stage K tile with RoPE
    {
      const int r  = tid >> 1;
      const int i0 = (tid & 1) * 16;
      const int pos = kt * 64 + r;
      const float* kp = qkv + (rowBase + pos) * NQKV_ + DI_ + h * DH_;
      #pragma unroll
      for (int ii = 0; ii < 16; ++ii) {
        int i = i0 + ii;
        float inv = __expf(-LN10000_OVER_32 * (float)i);
        float ang = (float)pos * inv;
        float cs = __cosf(ang), sn = __sinf(ang);
        float t1 = kp[2 * i], t2 = kp[2 * i + 1];
        Ks[r][2 * i]     = (_Float16)(t1 * cs - t2 * sn);
        Ks[r][2 * i + 1] = (_Float16)(t1 * sn + t2 * cs);
      }
    }
    // stage V tile transposed, with first_v store / mix interpolation
    {
      const int r  = tid >> 1;
      const int d0 = (tid & 1) * 32;
      const size_t row = rowBase + kt * 64 + r;
      const float* vp = qkv + row * NQKV_ + 2 * DI_ + h * DH_;
      const float mval = useMix ? mix[row * H_ + h] : 0.f;
      #pragma unroll
      for (int dd = 0; dd < 32; ++dd) {
        int d = d0 + dd;
        float v = vp[d];
        if (storeFirst) firstv[row * DI_ + h * DH_ + d] = v;
        if (useMix) {
          float fv = firstv[row * DI_ + h * DH_ + d];
          v = v + mval * (fv - v);
        }
        Vt[d][r] = (_Float16)v;
      }
    }
    __syncthreads();

    // scores: S = Q K^T  (16q x 64keys per wave)
    v8f sc[4] = {};
    #pragma unroll
    for (int nt = 0; nt < 4; ++nt) {
      #pragma unroll
      for (int kc = 0; kc < 2; ++kc) {
        V16 u;
        u.h[0] = *(const h8*)&Ks[nt * 16 + lm][kc * 32 + 8 * lh];
        u.h[1] = *(const h8*)&Ks[nt * 16 + lm][kc * 32 + 16 + 8 * lh];
        sc[nt] = __builtin_amdgcn_wmma_f32_16x16x32_f16(
            false, aQ[kc], false, u.v, (short)0, sc[nt], false, false);
      }
    }

    // block-causal mask (only the diagonal 64x64 tile can be partial)
    if (kt == qt) {
      #pragma unroll
      for (int nt = 0; nt < 4; ++nt) {
        int jg = kt * 64 + nt * 16 + lm;
        #pragma unroll
        for (int r = 0; r < 8; ++r) {
          int ig = q0 + wave * 16 + lh * 8 + r;
          if ((ig >> 5) < (jg >> 5)) sc[nt][r] = -3.402823466e38f;
        }
      }
    }

    // online softmax (row = lh*8+r, 16 lanes per row-group)
    #pragma unroll
    for (int r = 0; r < 8; ++r) {
      float rm = fmaxf(fmaxf(sc[0][r], sc[1][r]), fmaxf(sc[2][r], sc[3][r]));
      #pragma unroll
      for (int off = 1; off < 16; off <<= 1) rm = fmaxf(rm, __shfl_xor(rm, off, 32));
      float mnew  = fmaxf(mrow[r], rm);
      float alpha = __expf(mrow[r] - mnew);
      float rs = 0.f;
      #pragma unroll
      for (int nt = 0; nt < 4; ++nt) {
        float p = __expf(sc[nt][r] - mnew);
        sc[nt][r] = p;
        rs += p;
      }
      #pragma unroll
      for (int off = 1; off < 16; off <<= 1) rs += __shfl_xor(rs, off, 32);
      lrow[r] = lrow[r] * alpha + rs;
      mrow[r] = mnew;
      #pragma unroll
      for (int dt = 0; dt < 4; ++dt) acc[dt][r] *= alpha;
    }

    // P to per-wave LDS (C-layout -> A-layout relayout); same-wave LDS in-order
    #pragma unroll
    for (int nt = 0; nt < 4; ++nt)
      #pragma unroll
      for (int r = 0; r < 8; ++r)
        Ps[wave][lh * 8 + r][nt * 16 + lm] = (_Float16)sc[nt][r];

    // O += P @ V
    #pragma unroll
    for (int dt = 0; dt < 4; ++dt) {
      #pragma unroll
      for (int kc = 0; kc < 2; ++kc) {
        V16 ua, ub;
        ua.h[0] = *(const h8*)&Ps[wave][lm][kc * 32 + 8 * lh];
        ua.h[1] = *(const h8*)&Ps[wave][lm][kc * 32 + 16 + 8 * lh];
        ub.h[0] = *(const h8*)&Vt[dt * 16 + lm][kc * 32 + 8 * lh];
        ub.h[1] = *(const h8*)&Vt[dt * 16 + lm][kc * 32 + 16 + 8 * lh];
        acc[dt] = __builtin_amdgcn_wmma_f32_16x16x32_f16(
            false, ua.v, false, ub.v, (short)0, acc[dt], false, false);
      }
    }
  }

  #pragma unroll
  for (int dt = 0; dt < 4; ++dt) {
    #pragma unroll
    for (int r = 0; r < 8; ++r) {
      int qrow = q0 + wave * 16 + lh * 8 + r;
      o[(rowBase + qrow) * DI_ + h * DH_ + dt * 16 + lm] =
          (_Float16)(acc[dt][r] / lrow[r]);
    }
  }
}

// ---------------------------------------------------------------- launch ----
extern "C" void kernel_launch(void* const* d_in, const int* in_sizes, int n_in,
                              void* d_out, int out_size, void* d_ws, size_t ws_size,
                              hipStream_t stream) {
  (void)in_sizes; (void)n_in; (void)out_size; (void)ws_size;
  const float* tokens   = (const float*)d_in[0];
  const float* attn_nw  = (const float*)d_in[1];
  const float* qkv_w    = (const float*)d_in[2];
  const float* out_w    = (const float*)d_in[3];
  const float* mix_w    = (const float*)d_in[4];
  const float* mix_b    = (const float*)d_in[5];
  const float* ff_nw    = (const float*)d_in[6];
  const float* ff_in_w  = (const float*)d_in[7];
  const float* ff_in_b  = (const float*)d_in[8];
  const float* ff_out_w = (const float*)d_in[9];
  const float* ff_out_b = (const float*)d_in[10];
  const float* final_nw = (const float*)d_in[11];

  char* p = (char*)d_ws;
  auto allocF = [&](size_t n) { float* q = (float*)p;    p += n * 4; return q; };
  auto allocH = [&](size_t n) { _Float16* q = (_Float16*)p; p += n * 2; return q; };

  float*    t     = allocF((size_t)M_ * D_);
  float*    qkv   = allocF((size_t)M_ * NQKV_);
  float*    hbuf  = allocF((size_t)M_ * NFFP_);
  float*    fv    = allocF((size_t)M_ * DI_);
  float*    mix   = allocF((size_t)M_ * H_);
  float*    binp  = allocF((size_t)L_ * NFFP_);
  _Float16* xh    = allocH((size_t)M_ * D_);
  _Float16* oh    = allocH((size_t)M_ * D_);
  _Float16* wqkv  = allocH((size_t)L_ * NQKV_ * D_);
  _Float16* wout  = allocH((size_t)L_ * D_ * D_);
  _Float16* wffin = allocH((size_t)L_ * NFFP_ * D_);
  _Float16* wffout= allocH((size_t)L_ * D_ * KFFP_);
  _Float16* gh    = (_Float16*)qkv;   // alias: qkv dead once attention is done

  hipMemcpyAsync(t, tokens, sizeof(float) * (size_t)M_ * D_,
                 hipMemcpyDeviceToDevice, stream);

  // one-shot weight convert + transpose + zero-pad (deterministic every call)
  for (int l = 0; l < L_; ++l) {
    wcvt_kernel<<<(NQKV_ * D_) / 256, 256, 0, stream>>>(
        qkv_w + (size_t)l * D_ * NQKV_, wqkv + (size_t)l * NQKV_ * D_,
        D_, NQKV_, D_, NQKV_);
    wcvt_kernel<<<(D_ * D_) / 256, 256, 0, stream>>>(
        out_w + (size_t)l * DI_ * D_, wout + (size_t)l * D_ * D_,
        DI_, D_, DI_, D_);
    wcvt_kernel<<<(NFFP_ * D_) / 256, 256, 0, stream>>>(
        ff_in_w + (size_t)l * D_ * NFF_, wffin + (size_t)l * NFFP_ * D_,
        D_, NFF_, D_, NFFP_);
    wcvt_kernel<<<(D_ * KFFP_) / 256, 256, 0, stream>>>(
        ff_out_w + (size_t)l * DHID_ * D_, wffout + (size_t)l * D_ * KFFP_,
        DHID_, D_, KFFP_, D_);
    padb_kernel<<<(NFFP_ + 255) / 256, 256, 0, stream>>>(
        ff_in_b + (size_t)l * NFF_, binp + (size_t)l * NFFP_, NFF_, NFFP_);
  }

  for (int l = 0; l < L_; ++l) {
    rmsnorm_kernel<<<M_, 256, 0, stream>>>(t, attn_nw + l * D_, nullptr, xh);
    gemm_wmma<<<dim3(NQKV_ / 64, M_ / 128), 256, 0, stream>>>(
        xh, wqkv + (size_t)l * NQKV_ * D_, nullptr, qkv, D_, NQKV_, 0);
    if (l > 0)
      mix_kernel<<<(M_ * H_ + 255) / 256, 256, 0, stream>>>(
          xh, mix_w + (size_t)l * D_ * H_, mix_b + l * H_, mix);
    attn_kernel<<<dim3(S_ / 64, H_, B_), 128, 0, stream>>>(
        qkv, mix, fv, (l == 0) ? 1 : 0, (l > 0) ? 1 : 0, oh);
    gemm_wmma<<<dim3(D_ / 64, M_ / 128), 256, 0, stream>>>(
        oh, wout + (size_t)l * D_ * D_, nullptr, t, D_, D_, 1);
    rmsnorm_kernel<<<M_, 256, 0, stream>>>(t, ff_nw + l * D_, nullptr, xh);
    gemm_wmma<<<dim3(NFFP_ / 64, M_ / 128), 256, 0, stream>>>(
        xh, wffin + (size_t)l * NFFP_ * D_, binp + (size_t)l * NFFP_, hbuf,
        D_, NFFP_, 0);
    long long ne = (long long)M_ * KFFP_;
    glu_kernel<<<(unsigned)((ne + 255) / 256), 256, 0, stream>>>(hbuf, gh);
    gemm_wmma<<<dim3(D_ / 64, M_ / 128), 256, 0, stream>>>(
        gh, wffout + (size_t)l * D_ * KFFP_, ff_out_b + (size_t)l * D_, t,
        KFFP_, D_, 1);
  }
  rmsnorm_kernel<<<M_, 256, 0, stream>>>(t, final_nw, (float*)d_out, nullptr);
}
